// GemmaAttention_46308337385834
// MI455X (gfx1250) — matrix operand order repaired and checked by reading, not compiled, not run on hardware
//
#include <hip/hip_runtime.h>
#include <hip/hip_bf16.h>

// ---------------------------------------------------------------------------
// GemmaAttention forward for MI455X (gfx1250), wave32 + WMMA f32_16x16x32_f16
// GEMMs stage B tiles into LDS via gfx1250 async loads (ASYNCcnt) when the
// toolchain exposes the builtins; otherwise fall back to manual LDS staging.
// ---------------------------------------------------------------------------

typedef _Float16 half_t;
typedef half_t v8h  __attribute__((ext_vector_type(8)));
typedef half_t v16h __attribute__((ext_vector_type(16)));
typedef float  v8f  __attribute__((ext_vector_type(8)));
typedef int    v4i  __attribute__((ext_vector_type(4)));

#define NEG_BIG   (-1000000000.0f)
#define ATT_SCALE (0.0625f)          // 1/sqrt(256)

#if __has_builtin(__builtin_amdgcn_global_load_async_to_lds_b128)
#define HAS_ASYNC_LDS 1
#else
#define HAS_ASYNC_LDS 0
#endif

// Builtin parameter types (from hipcc diagnostic): v4i in addrspace(1) / (3).
typedef __attribute__((address_space(1))) v4i* gv4i_t;
typedef __attribute__((address_space(3))) v4i* lv4i_t;

__device__ __forceinline__ void wait_asynccnt0() {
#if __has_builtin(__builtin_amdgcn_s_wait_asynccnt)
    __builtin_amdgcn_s_wait_asynccnt(0);
#else
    asm volatile("s_wait_asynccnt 0" ::: "memory");
#endif
}

__device__ __forceinline__ v16h cat8(v8h lo, v8h hi) {
    return __builtin_shufflevector(lo, hi, 0,1,2,3,4,5,6,7,8,9,10,11,12,13,14,15);
}
__device__ __forceinline__ v16h ldg16(const half_t* p) { return *(const v16h*)p; }
__device__ __forceinline__ v8h  ldg8 (const half_t* p) { return *(const v8h*)p;  }

__device__ __forceinline__ v8f wmma_f16(v16h a, v16h b, v8f c) {
    // D = A(16x32 f16) * B(32x16 f16) + C(16x16 f32)
    return __builtin_amdgcn_wmma_f32_16x16x32_f16(
        /*neg_a=*/false, a, /*neg_b=*/false, b,
        /*c_mod=*/(short)0, c, /*reuse_a=*/false, /*reuse_b=*/false);
}

// ---------------------------------------------------------------------------
// Elementwise converts
// ---------------------------------------------------------------------------
__global__ void f32_to_f16_kernel(const float* __restrict__ src,
                                  half_t* __restrict__ dst, size_t n) {
    size_t i = (size_t)blockIdx.x * blockDim.x + threadIdx.x;
    if (i < n) dst[i] = (half_t)src[i];
}

// dst[(rowOff+n)*dstK + k] = src[k*N + n]   (transpose + convert)
__global__ void convT_kernel(const float* __restrict__ src,
                             half_t* __restrict__ dst,
                             int K, int N, int rowOff, int dstK) {
    size_t i = (size_t)blockIdx.x * blockDim.x + threadIdx.x;
    size_t total = (size_t)K * N;
    if (i >= total) return;
    int n = (int)(i % N);
    int k = (int)(i / N);
    dst[(size_t)(rowOff + n) * dstK + k] = (half_t)src[i];
}

// ---------------------------------------------------------------------------
// GEMM: C (MxN f32) = A (MxK f16, row-major) * BT (NxK f16, row-major = B^T)
// 128 threads = 4 waves. WG tile 128x64; each wave computes a 32x64 strip.
// B tile (64 rows x 32 k halves = 4 KB) is staged into LDS, double-buffered,
// via gfx1250 async-to-LDS loads so the DMA overlaps WMMA issue.
// ---------------------------------------------------------------------------
__global__ void __launch_bounds__(128)
gemm_wmma_kernel(const half_t* __restrict__ A, const half_t* __restrict__ BT,
                 float* __restrict__ C, int M, int N, int K) {
    __shared__ half_t lds_b[2][64][32];   // [buf][bt-row][k-half]

    const int lane = threadIdx.x & 31;
    const int wave = threadIdx.x >> 5;
    const int l16  = lane & 15;
    const int hs   = lane >> 4;            // 0 for lanes 0-15, 1 for 16-31
    const int m0   = blockIdx.y * 128 + wave * 32;
    const int n0   = blockIdx.x * 64;

    const half_t* arow0 = A + (size_t)(m0 + l16) * K;        // rows m0..m0+15
    const half_t* arow1 = A + (size_t)(m0 + 16 + l16) * K;   // rows m0+16..31

    v8f acc[2][4];
    #pragma unroll
    for (int mt = 0; mt < 2; ++mt)
        #pragma unroll
        for (int nt = 0; nt < 4; ++nt) {
            v8f z = {0.f,0.f,0.f,0.f,0.f,0.f,0.f,0.f};
            acc[mt][nt] = z;
        }

    // Stage one 64x32 BT tile into lds_b[buf]; 128 threads x 2 x 16B = 4 KB.
    auto stage_b = [&](int buf, int k0) {
        #pragma unroll
        for (int j = 0; j < 2; ++j) {
            int c   = threadIdx.x * 2 + j;   // 0..255 16-byte chunks
            int row = c >> 2;                // 0..63
            int off = (c & 3) * 8;           // half offset within 32
            const half_t* g = BT + (size_t)(n0 + row) * K + k0 + off;
            half_t*       l = &lds_b[buf][row][off];
#if HAS_ASYNC_LDS
            __builtin_amdgcn_global_load_async_to_lds_b128(
                (gv4i_t)g, (lv4i_t)l, 0, 0);
#else
            *(v8h*)l = ldg8(g);
#endif
        }
    };

    stage_b(0, 0);
    for (int k0 = 0; k0 < K; k0 += 32) {
        const int cur = (k0 >> 5) & 1;
#if HAS_ASYNC_LDS
        wait_asynccnt0();
#endif
        __syncthreads();                       // lds_b[cur] ready, readers done
        if (k0 + 32 < K) stage_b(1 - cur, k0 + 32);

        // A fragments (contiguous k, matching B): lanes 0-15 -> k {0..15},
        //                                         lanes 16-31 -> k {16..31}
        v16h a0 = cat8(ldg8(arow0 + k0 + 16 * hs), ldg8(arow0 + k0 + 8 + 16 * hs));
        v16h a1 = cat8(ldg8(arow1 + k0 + 16 * hs), ldg8(arow1 + k0 + 8 + 16 * hs));

        #pragma unroll
        for (int nt = 0; nt < 4; ++nt) {
            // B fragment from LDS: lane = output column, halves = 16 k values
            v16h b = *(const v16h*)&lds_b[cur][nt * 16 + l16][16 * hs];
            acc[0][nt] = wmma_f16(a0, b, acc[0][nt]);
            acc[1][nt] = wmma_f16(a1, b, acc[1][nt]);
        }
    }

    // C layout: elem r of accumulator = C[row + r + 8*hs][n0 + nt*16 + l16]
    #pragma unroll
    for (int mt = 0; mt < 2; ++mt)
        #pragma unroll
        for (int nt = 0; nt < 4; ++nt)
            #pragma unroll
            for (int r = 0; r < 8; ++r) {
                int row = m0 + mt * 16 + r + 8 * hs;
                int col = n0 + nt * 16 + l16;
                C[(size_t)row * N + col] = acc[mt][nt][r];
            }
}

// ---------------------------------------------------------------------------
// RoPE: read fused qkv f32 [B*S][2560], write Q f16 [B][8][S][256] and
// K f16 [B][S][256].  head index 8 == the single KV head's K.
// ---------------------------------------------------------------------------
__global__ void rope_kernel(const float* __restrict__ qkv,
                            half_t* __restrict__ Qo, half_t* __restrict__ Ko,
                            int B, int S) {
    size_t idx = (size_t)blockIdx.x * blockDim.x + threadIdx.x;
    size_t total = (size_t)B * S * 9 * 128;
    if (idx >= total) return;
    int i  = (int)(idx % 128);           // rotary pair index
    size_t t = idx / 128;
    int hh = (int)(t % 9);  t /= 9;
    int s  = (int)(t % S);
    int b  = (int)(t / S);

    float invf = __powf(10000.0f, -(float)i * (1.0f / 128.0f));
    float ang  = (float)s * invf;
    float c, sn;
    __sincosf(ang, &sn, &c);

    const float* row = qkv + ((size_t)b * S + s) * 2560;
    int col = (hh < 8) ? (hh * 256 + i) : (2048 + i);
    float x1 = row[col];
    float x2 = row[col + 128];
    float o1 = x1 * c - x2 * sn;
    float o2 = x2 * c + x1 * sn;

    if (hh < 8) {
        half_t* dst = Qo + (((size_t)(b * 8 + hh)) * S + s) * 256 + i;
        dst[0]   = (half_t)o1;
        dst[128] = (half_t)o2;
    } else {
        half_t* dst = Ko + ((size_t)b * S + s) * 256 + i;
        dst[0]   = (half_t)o1;
        dst[128] = (half_t)o2;
    }
}

// V transpose: VT[b][d][s] = (f16) qkv[b*S+s][2304 + d]
__global__ void vtrans_kernel(const float* __restrict__ qkv,
                              half_t* __restrict__ VT, int B, int S) {
    size_t idx = (size_t)blockIdx.x * blockDim.x + threadIdx.x;
    size_t total = (size_t)B * S * 256;
    if (idx >= total) return;
    int d = (int)(idx % 256);
    size_t t = idx / 256;
    int s = (int)(t % S);
    int b = (int)(t / S);
    VT[((size_t)b * 256 + d) * S + s] =
        (half_t)qkv[((size_t)b * S + s) * 2560 + 2304 + d];
}

// ---------------------------------------------------------------------------
// Flash attention (causal, GQA: 1 KV head shared by 8 Q heads).
// grid = (S/64, 8, B), block = 128 (4 waves).  Each wave: 16 query rows,
// Q resident in registers, streams K/V in 32-key blocks with online softmax.
// ---------------------------------------------------------------------------
__global__ void __launch_bounds__(128)
attn_kernel(const half_t* __restrict__ Q,   // [B][8][S][256]
            const half_t* __restrict__ Kc,  // [B][S][256]
            const half_t* __restrict__ VT,  // [B][256][S]
            half_t* __restrict__ O,         // [B*S][2048]
            int S) {
    __shared__ half_t lds_p[4][16][32];     // per-wave P tile (16x32 f16)

    const int lane = threadIdx.x & 31;
    const int wave = threadIdx.x >> 5;
    const int l16  = lane & 15;
    const int hs   = lane >> 4;
    const int b    = blockIdx.z;
    const int h    = blockIdx.y;
    const int q0   = blockIdx.x * 64 + wave * 16;

    // Load this wave's 16x256 Q strip as 8 A-fragments (resident).
    const half_t* qrow = Q + (((size_t)(b * 8 + h)) * S + q0 + l16) * 256;
    v16h qa[8];
    #pragma unroll
    for (int ks = 0; ks < 8; ++ks) {
        v8h lo = ldg8(qrow + ks * 32 + 16 * hs);
        v8h hi = ldg8(qrow + ks * 32 + 8 + 16 * hs);
        qa[ks] = cat8(lo, hi);
    }

    float m[8], l[8];
    v8f acc[16];
    #pragma unroll
    for (int r = 0; r < 8; ++r) { m[r] = -1e30f; l[r] = 0.f; }
    #pragma unroll
    for (int nt = 0; nt < 16; ++nt) {
        v8f z = {0.f,0.f,0.f,0.f,0.f,0.f,0.f,0.f};
        acc[nt] = z;
    }

    const int kend = blockIdx.x * 64 + 64;   // WG-uniform (causal bound)
    const half_t* vbase = VT + (size_t)b * 256 * S;

    for (int kb = 0; kb < kend; kb += 32) {
        // ---- scores: S(16x32) = Q(16x256) * K^T, two 16x16 tiles ----
        v8f s0 = {0.f,0.f,0.f,0.f,0.f,0.f,0.f,0.f};
        v8f s1 = s0;
        const half_t* kbase = Kc + ((size_t)b * S + kb) * 256;
        #pragma unroll
        for (int ks = 0; ks < 8; ++ks) {
            v16h b0 = ldg16(kbase + (size_t)l16 * 256        + ks * 32 + 16 * hs);
            v16h b1 = ldg16(kbase + (size_t)(16 + l16) * 256 + ks * 32 + 16 * hs);
            s0 = wmma_f16(qa[ks], b0, s0);
            s1 = wmma_f16(qa[ks], b1, s1);
        }

        // ---- online softmax (rows striped across lanes 0-15 / 16-31) ----
        float p0[8], p1[8], alpha[8];
        #pragma unroll
        for (int r = 0; r < 8; ++r) {
            int row = q0 + r + 8 * hs;
            float x0 = s0[r] * ATT_SCALE + ((kb + l16)      <= row ? 0.f : NEG_BIG);
            float x1 = s1[r] * ATT_SCALE + ((kb + 16 + l16) <= row ? 0.f : NEG_BIG);
            float bm = fmaxf(x0, x1);
            #pragma unroll
            for (int t = 1; t < 16; t <<= 1)
                bm = fmaxf(bm, __shfl_xor(bm, t));
            float mn = fmaxf(m[r], bm);
            alpha[r] = __expf(m[r] - mn);
            p0[r] = __expf(x0 - mn);
            p1[r] = __expf(x1 - mn);
            float rs = p0[r] + p1[r];
            #pragma unroll
            for (int t = 1; t < 16; t <<= 1)
                rs += __shfl_xor(rs, t);
            l[r] = l[r] * alpha[r] + rs;
            m[r] = mn;
        }
        #pragma unroll
        for (int nt = 0; nt < 16; ++nt)
            #pragma unroll
            for (int r = 0; r < 8; ++r)
                acc[nt][r] *= alpha[r];

        // ---- re-layout P through LDS into WMMA A-fragment format ----
        #pragma unroll
        for (int r = 0; r < 8; ++r) {
            int row = r + 8 * hs;
            lds_p[wave][row][l16]      = (half_t)p0[r];
            lds_p[wave][row][16 + l16] = (half_t)p1[r];
        }
        __syncthreads();
        v8h plo = *(const v8h*)&lds_p[wave][l16][16 * hs];
        v8h phi = *(const v8h*)&lds_p[wave][l16][8 + 16 * hs];
        v16h pa = cat8(plo, phi);

        // ---- O(16x256) += P(16x32) * V(32x256) : 16 WMMAs ----
        #pragma unroll
        for (int nt = 0; nt < 16; ++nt) {
            v16h bv = ldg16(vbase + (size_t)(nt * 16 + l16) * S + kb + 16 * hs);
            acc[nt] = wmma_f16(pa, bv, acc[nt]);
        }
        __syncthreads();
    }

    // ---- epilogue: normalize and store fp16 ----
    float invl[8];
    #pragma unroll
    for (int r = 0; r < 8; ++r) invl[r] = 1.0f / l[r];
    #pragma unroll
    for (int nt = 0; nt < 16; ++nt)
        #pragma unroll
        for (int r = 0; r < 8; ++r) {
            int row = q0 + r + 8 * hs;
            O[((size_t)b * S + row) * 2048 + h * 256 + nt * 16 + l16] =
                (half_t)(acc[nt][r] * invl[r]);
        }
}

// ---------------------------------------------------------------------------
// Host-side orchestration
// ---------------------------------------------------------------------------
extern "C" void kernel_launch(void* const* d_in, const int* in_sizes, int n_in,
                              void* d_out, int out_size, void* d_ws, size_t ws_size,
                              hipStream_t stream) {
    (void)in_sizes; (void)n_in; (void)out_size; (void)ws_size;
    const float* hidden = (const float*)d_in[0];
    // d_in[1] = attention_mask (causal; implemented analytically)
    // d_in[2] = position_ids   (arange; implemented analytically)
    const float* Wq = (const float*)d_in[3];
    const float* Wk = (const float*)d_in[4];
    const float* Wv = (const float*)d_in[5];
    const float* Wo = (const float*)d_in[6];
    float* out = (float*)d_out;

    const int B = 2, S = 2048, H = 2048;
    const int M = B * S;                 // 4096
    const int NQKV = 2560;               // 8*256 + 256 + 256

    char* ws = (char*)d_ws;
    size_t off = 0;
    auto alloc = [&](size_t bytes) {
        void* p = ws + off;
        off += (bytes + 255) & ~(size_t)255;
        return p;
    };
    half_t* Xh     = (half_t*)alloc((size_t)M * H * 2);        // hidden f16
    half_t* WqkvT  = (half_t*)alloc((size_t)NQKV * H * 2);     // [2560][2048]
    half_t* WoT    = (half_t*)alloc((size_t)H * 2048 * 2);     // [2048][2048]
    float*  qkv    = (float*) alloc((size_t)M * NQKV * 4);     // fused QKV f32
    half_t* Qf     = (half_t*)alloc((size_t)B * 8 * S * 256 * 2);
    half_t* Kf     = (half_t*)alloc((size_t)B * S * 256 * 2);
    half_t* VT     = (half_t*)alloc((size_t)B * 256 * S * 2);
    half_t* AO     = (half_t*)alloc((size_t)M * 2048 * 2);     // attn out f16

    // 1) converts
    {
        size_t n = (size_t)M * H;
        f32_to_f16_kernel<<<(unsigned)((n + 255) / 256), 256, 0, stream>>>(hidden, Xh, n);
    }
    {
        size_t n = (size_t)H * 2048;
        convT_kernel<<<(unsigned)((n + 255) / 256), 256, 0, stream>>>(Wq, WqkvT, H, 2048, 0, H);
    }
    {
        size_t n = (size_t)H * 256;
        convT_kernel<<<(unsigned)((n + 255) / 256), 256, 0, stream>>>(Wk, WqkvT, H, 256, 2048, H);
        convT_kernel<<<(unsigned)((n + 255) / 256), 256, 0, stream>>>(Wv, WqkvT, H, 256, 2304, H);
    }
    {
        size_t n = (size_t)H * 2048;
        convT_kernel<<<(unsigned)((n + 255) / 256), 256, 0, stream>>>(Wo, WoT, 2048, 2048, 0, 2048);
    }

    // 2) fused QKV projection (WMMA GEMM, async-LDS double-buffered B)
    gemm_wmma_kernel<<<dim3(NQKV / 64, M / 128), 128, 0, stream>>>(Xh, WqkvT, qkv, M, NQKV, H);

    // 3) RoPE + V transpose
    {
        size_t n = (size_t)B * S * 9 * 128;
        rope_kernel<<<(unsigned)((n + 255) / 256), 256, 0, stream>>>(qkv, Qf, Kf, B, S);
    }
    {
        size_t n = (size_t)B * S * 256;
        vtrans_kernel<<<(unsigned)((n + 255) / 256), 256, 0, stream>>>(qkv, VT, B, S);
    }

    // 4) causal flash attention (WMMA)
    attn_kernel<<<dim3(S / 64, 8, B), 128, 0, stream>>>(Qf, Kf, VT, AO, S);

    // 5) output projection (WMMA GEMM) -> fp32 d_out
    gemm_wmma_kernel<<<dim3(2048 / 64, M / 128), 128, 0, stream>>>(AO, WoT, out, M, 2048, 2048);
}
